// GAT_Encoder_4690104287830
// MI455X (gfx1250) — matrix operand order repaired and checked by reading, not compiled, not run on hardware
//
#include <hip/hip_runtime.h>
#include <stdint.h>

// ---------------- types for WMMA (gfx1250, wave32) ----------------
typedef __attribute__((ext_vector_type(16))) __bf16 v16bf;
typedef __attribute__((ext_vector_type(8)))  float  v8f;

struct U128 { unsigned long long a, b; };        // POD 16B chunk
union AFrag { v16bf v; U128 q[2]; };             // A-fragment loader

#define ATT_SLOPE 0.2f
#define ACT_SLOPE 0.01f

__device__ __forceinline__ float lrelu(float x, float s) { return x > 0.f ? x : s * x; }

// ---------------- f32 -> bf16 convert ----------------
__global__ void k_f32_to_bf16(const float* __restrict__ in, __bf16* __restrict__ out, size_t n) {
  size_t i = (size_t)blockIdx.x * blockDim.x + threadIdx.x;
  if (i < n) out[i] = (__bf16)in[i];
}

// ---------------- pack W[K,M] (row-major f32) into WMMA B-fragment tile order ----------------
// Tile (kt,nt) covers K in [kt*32,kt*32+32), N in [nt*16,nt*16+16).
// B layout (16-bit, 32x16): lane L element j holds B[8*(L/16)+(j<8?j:j+8)][L%16].
__global__ void k_pack_w(const float* __restrict__ W, __bf16* __restrict__ Wp,
                         int K, int M, int ntilesPad) {
  int ktiles = K / 32;
  int total = ktiles * ntilesPad * 32;
  int t = blockIdx.x * blockDim.x + threadIdx.x;
  if (t >= total) return;
  int lane = t & 31, tile = t >> 5;
  int nt = tile % ntilesPad, kt = tile / ntilesPad;
  int col = nt * 16 + (lane & 15);
  int kb  = kt * 32 + ((lane >> 4) << 3);
  __bf16* dst = Wp + (size_t)tile * 512 + lane * 16;
#pragma unroll
  for (int j = 0; j < 16; ++j) {
    float v = (col < M) ? W[(size_t)(kb + (j < 8 ? j : j + 8)) * M + col] : 0.f;
    dst[j] = (__bf16)v;
  }
}

// ---------------- WMMA GEMM: out[rows,M] = A[rows,K](bf16) @ W + bias ----------------
// One wave computes a 16x64 output tile (4 accumulators), K stepped by 32.
__global__ void k_gemm_wmma(const __bf16* __restrict__ A, const __bf16* __restrict__ Wp,
                            const float* __restrict__ bias, float* __restrict__ out,
                            int rows, int K, int M, int ntilesPad) {
  int wid = threadIdx.x >> 5, lane = threadIdx.x & 31;
  int mtile = blockIdx.x * (blockDim.x >> 5) + wid;
  if (mtile * 16 >= rows) return;
  int half = lane >> 4, l16 = lane & 15;
  int ntg = blockIdx.y;                       // group of 4 column tiles (64 cols)
  const __bf16* arow = A + (size_t)(mtile * 16 + l16) * K;
  v8f acc0 = {}, acc1 = {}, acc2 = {}, acc3 = {};
  int ktiles = K >> 5;
  for (int kt = 0; kt < ktiles; ++kt) {
    AFrag a;
    a.q[0] = *(const U128*)(arow + kt * 32 + 8 * half);
    a.q[1] = *(const U128*)(arow + kt * 32 + 16 + 8 * half);
    const __bf16* bbase = Wp + ((size_t)kt * ntilesPad + ntg * 4) * 512 + lane * 16;
    v16bf b0 = *(const v16bf*)(bbase);
    v16bf b1 = *(const v16bf*)(bbase + 512);
    v16bf b2 = *(const v16bf*)(bbase + 1024);
    v16bf b3 = *(const v16bf*)(bbase + 1536);
    acc0 = __builtin_amdgcn_wmma_f32_16x16x32_bf16(false, a.v, false, b0, (short)0, acc0, false, false);
    acc1 = __builtin_amdgcn_wmma_f32_16x16x32_bf16(false, a.v, false, b1, (short)0, acc1, false, false);
    acc2 = __builtin_amdgcn_wmma_f32_16x16x32_bf16(false, a.v, false, b2, (short)0, acc2, false, false);
    acc3 = __builtin_amdgcn_wmma_f32_16x16x32_bf16(false, a.v, false, b3, (short)0, acc3, false, false);
  }
  auto store_tile = [&](int t, v8f ac) {
    int col = (ntg * 4 + t) * 16 + l16;
    if (col < M) {
      float bv = bias[col];
      float* orow = out + (size_t)(mtile * 16 + 8 * half) * M + col; // D: VGPR i -> row i+8*half
#pragma unroll
      for (int i = 0; i < 8; ++i) orow[(size_t)i * M] = ac[i] + bv;
    }
  };
  store_tile(0, acc0); store_tile(1, acc1); store_tile(2, acc2); store_tile(3, acc3);
}

// ---------------- CSR build ----------------
__global__ void k_count(const int* __restrict__ dst, int* __restrict__ counts, int E_) {
  int e = blockIdx.x * blockDim.x + threadIdx.x;
  if (e < E_) atomicAdd(&counts[dst[e]], 1);
}
__global__ void k_scan(const int* __restrict__ counts, int* __restrict__ rowptr, int n) {
  __shared__ int buf[1024];
  __shared__ int carry;
  int tid = threadIdx.x;
  if (tid == 0) carry = 0;
  __syncthreads();
  for (int base = 0; base < n; base += 1024) {
    int i = base + tid;
    int v = (i < n) ? counts[i] : 0;
    buf[tid] = v; __syncthreads();
    for (int off = 1; off < 1024; off <<= 1) {
      int t = (tid >= off) ? buf[tid - off] : 0;
      __syncthreads();
      buf[tid] += t;
      __syncthreads();
    }
    int c0 = carry;
    if (i < n) rowptr[i] = c0 + buf[tid] - v;  // exclusive
    __syncthreads();
    if (tid == 0) carry = c0 + buf[1023];
    __syncthreads();
  }
  if (tid == 0) rowptr[n] = carry;
}
__global__ void k_copy_int(const int* __restrict__ in, int* __restrict__ out, int n) {
  int i = blockIdx.x * blockDim.x + threadIdx.x;
  if (i < n) out[i] = in[i];
}
__global__ void k_scatter(const int* __restrict__ dst, int* __restrict__ cursor,
                          int* __restrict__ csr, int E_) {
  int e = blockIdx.x * blockDim.x + threadIdx.x;
  if (e < E_) { int p = atomicAdd(&cursor[dst[e]], 1); csr[p] = e; }
}

// ---------------- edge logits, HC=256/C=64 float4 path: wave per edge ----------------
// lane covers channels [4*lane,4*lane+4) + r*128; head = 2r + lane/16 (uniform per chunk).
__global__ void k_edge_logits_v4(const float* __restrict__ xl, const float* __restrict__ xr,
                                 const float* __restrict__ att,
                                 const int* __restrict__ src, const int* __restrict__ dst,
                                 float* __restrict__ logits, int E_) {
  int e = blockIdx.x * (blockDim.x >> 5) + (threadIdx.x >> 5);
  if (e >= E_) return;
  int lane = threadIdx.x & 31;
  const float* pl = xl + (size_t)src[e] * 256;
  const float* pr = xr + (size_t)dst[e] * 256;
  float s0 = 0.f, s1 = 0.f, s2 = 0.f, s3 = 0.f;
  bool lo = (lane < 16);
#pragma unroll
  for (int r = 0; r < 2; ++r) {
    int off = r * 128 + 4 * lane;
    float4 vl = *(const float4*)(pl + off);
    float4 vr = *(const float4*)(pr + off);
    float4 va = *(const float4*)(att + off);
    float p = lrelu(vl.x + vr.x, ATT_SLOPE) * va.x
            + lrelu(vl.y + vr.y, ATT_SLOPE) * va.y
            + lrelu(vl.z + vr.z, ATT_SLOPE) * va.z
            + lrelu(vl.w + vr.w, ATT_SLOPE) * va.w;
    if (r == 0) { s0 += lo ? p : 0.f; s1 += lo ? 0.f : p; }
    else        { s2 += lo ? p : 0.f; s3 += lo ? 0.f : p; }
  }
#pragma unroll
  for (int off = 16; off > 0; off >>= 1) {
    s0 += __shfl_xor(s0, off, 32);
    s1 += __shfl_xor(s1, off, 32);
    s2 += __shfl_xor(s2, off, 32);
    s3 += __shfl_xor(s3, off, 32);
  }
  if (lane == 0) *(float4*)(logits + (size_t)e * 4) = make_float4(s0, s1, s2, s3);
}

// ---------------- edge logits, generic scalar (used for latent layer HC=40,C=10) ----------------
template<int HC, int C>
__global__ void k_edge_logits(const float* __restrict__ xl, const float* __restrict__ xr,
                              const float* __restrict__ att,
                              const int* __restrict__ src, const int* __restrict__ dst,
                              float* __restrict__ logits, int E_) {
  int e = blockIdx.x * (blockDim.x >> 5) + (threadIdx.x >> 5);
  if (e >= E_) return;
  int lane = threadIdx.x & 31;
  const float* pl = xl + (size_t)src[e] * HC;
  const float* pr = xr + (size_t)dst[e] * HC;
  float s0 = 0.f, s1 = 0.f, s2 = 0.f, s3 = 0.f;
  constexpr int R = (HC + 31) / 32;
#pragma unroll
  for (int r = 0; r < R; ++r) {
    int idx = lane + 32 * r;
    if (HC % 32 == 0 || idx < HC) {
      float v = pl[idx] + pr[idx];
      v = lrelu(v, ATT_SLOPE) * att[idx];
      int h = idx / C;
      s0 += (h == 0) ? v : 0.f;
      s1 += (h == 1) ? v : 0.f;
      s2 += (h == 2) ? v : 0.f;
      s3 += (h == 3) ? v : 0.f;
    }
  }
#pragma unroll
  for (int off = 16; off > 0; off >>= 1) {
    s0 += __shfl_xor(s0, off, 32);
    s1 += __shfl_xor(s1, off, 32);
    s2 += __shfl_xor(s2, off, 32);
    s3 += __shfl_xor(s3, off, 32);
  }
  if (lane == 0) *(float4*)(logits + (size_t)e * 4) = make_float4(s0, s1, s2, s3);
}

// ---------------- per-node softmax stats via CSR (all 4 heads as float4, no atomics) ----------------
__global__ void k_softmax_stats(const float* __restrict__ logits, const int* __restrict__ rowptr,
                                const int* __restrict__ csr, float* __restrict__ mbuf,
                                float* __restrict__ dbuf, int n) {
  int node = blockIdx.x * blockDim.x + threadIdx.x;
  if (node >= n) return;
  int jb = rowptr[node], je = rowptr[node + 1];
  float ninf = -__builtin_inff();
  float4 m = make_float4(ninf, ninf, ninf, ninf);
  for (int j = jb; j < je; ++j) {
    float4 lv = *(const float4*)(logits + (size_t)csr[j] * 4);
    m.x = fmaxf(m.x, lv.x); m.y = fmaxf(m.y, lv.y);
    m.z = fmaxf(m.z, lv.z); m.w = fmaxf(m.w, lv.w);
  }
  float4 s = make_float4(0.f, 0.f, 0.f, 0.f);
  for (int j = jb; j < je; ++j) {
    float4 lv = *(const float4*)(logits + (size_t)csr[j] * 4);
    s.x += __expf(lv.x - m.x); s.y += __expf(lv.y - m.y);
    s.z += __expf(lv.z - m.z); s.w += __expf(lv.w - m.w);
  }
  *(float4*)(mbuf + (size_t)node * 4) = m;
  *(float4*)(dbuf + (size_t)node * 4) = s;
}

// ---------------- aggregation HC=256/C=64 float4 path: wave per node; no float atomics ----------------
__global__ void k_aggregate_v4(const float* __restrict__ xl, const float* __restrict__ logits,
                               const float* __restrict__ mbuf, const float* __restrict__ dbuf,
                               const int* __restrict__ rowptr, const int* __restrict__ csr,
                               const int* __restrict__ src,
                               float* __restrict__ alpha_out, float* __restrict__ acc, int n) {
  int w = blockIdx.x * (blockDim.x >> 5) + (threadIdx.x >> 5);
  if (w >= n) return;
  int lane = threadIdx.x & 31;
  bool lo = (lane < 16);
  float4 acc0 = make_float4(0.f, 0.f, 0.f, 0.f);
  float4 acc1 = make_float4(0.f, 0.f, 0.f, 0.f);
  float mh = 0.f, dh = 1.f;
  if (lane < 4) { mh = mbuf[w * 4 + lane]; dh = dbuf[w * 4 + lane]; }
  int jb = rowptr[w], je = rowptr[w + 1];
  for (int j = jb; j < je; ++j) {
    int e = csr[j];
    float al = 0.f;
    if (lane < 4) {
      al = __expf(logits[(size_t)e * 4 + lane] - mh) / dh;
      alpha_out[(size_t)e * 4 + lane] = al;
    }
    float a0 = __shfl(al, 0, 32), a1 = __shfl(al, 1, 32);
    float a2 = __shfl(al, 2, 32), a3 = __shfl(al, 3, 32);
    float aw0 = lo ? a0 : a1;         // head for chunk r=0 is lane<16 ? 0 : 1
    float aw1 = lo ? a2 : a3;         // head for chunk r=1 is lane<16 ? 2 : 3
    const float* ps = xl + (size_t)src[e] * 256;
    float4 v0 = *(const float4*)(ps + 4 * lane);
    float4 v1 = *(const float4*)(ps + 128 + 4 * lane);
    acc0.x = fmaf(aw0, v0.x, acc0.x); acc0.y = fmaf(aw0, v0.y, acc0.y);
    acc0.z = fmaf(aw0, v0.z, acc0.z); acc0.w = fmaf(aw0, v0.w, acc0.w);
    acc1.x = fmaf(aw1, v1.x, acc1.x); acc1.y = fmaf(aw1, v1.y, acc1.y);
    acc1.z = fmaf(aw1, v1.z, acc1.z); acc1.w = fmaf(aw1, v1.w, acc1.w);
  }
  float* pa = acc + (size_t)w * 256;
  *(float4*)(pa + 4 * lane) = acc0;
  *(float4*)(pa + 128 + 4 * lane) = acc1;
}

// ---------------- aggregation generic scalar (latent layer HC=40,C=10) ----------------
template<int HC, int C>
__global__ void k_aggregate(const float* __restrict__ xl, const float* __restrict__ logits,
                            const float* __restrict__ mbuf, const float* __restrict__ dbuf,
                            const int* __restrict__ rowptr, const int* __restrict__ csr,
                            const int* __restrict__ src,
                            float* __restrict__ alpha_out, float* __restrict__ acc, int n) {
  int w = blockIdx.x * (blockDim.x >> 5) + (threadIdx.x >> 5);
  if (w >= n) return;
  int lane = threadIdx.x & 31;
  constexpr int R = (HC + 31) / 32;
  float a[R];
#pragma unroll
  for (int r = 0; r < R; ++r) a[r] = 0.f;
  float mh = 0.f, dh = 1.f;
  if (lane < 4) { mh = mbuf[w * 4 + lane]; dh = dbuf[w * 4 + lane]; }
  int jb = rowptr[w], je = rowptr[w + 1];
  for (int j = jb; j < je; ++j) {
    int e = csr[j];
    float al = 0.f;
    if (lane < 4) {
      al = __expf(logits[(size_t)e * 4 + lane] - mh) / dh;
      alpha_out[(size_t)e * 4 + lane] = al;
    }
    float a0 = __shfl(al, 0, 32), a1 = __shfl(al, 1, 32);
    float a2 = __shfl(al, 2, 32), a3 = __shfl(al, 3, 32);
    const float* ps = xl + (size_t)src[e] * HC;
#pragma unroll
    for (int r = 0; r < R; ++r) {
      int idx = lane + 32 * r;
      if (HC % 32 == 0 || idx < HC) {
        int h = idx / C;
        float aw = (h == 0) ? a0 : ((h == 1) ? a1 : ((h == 2) ? a2 : a3));
        a[r] = fmaf(aw, ps[idx], a[r]);
      }
    }
  }
#pragma unroll
  for (int r = 0; r < R; ++r) {
    int idx = lane + 32 * r;
    if (HC % 32 == 0 || idx < HC) acc[(size_t)w * HC + idx] = a[r];
  }
}

// ---------------- finalize ----------------
__global__ void k_fin_concat4(const float* __restrict__ acc, const float* __restrict__ bias,
                              float* __restrict__ out, int n, int M) {
  size_t i = (size_t)blockIdx.x * blockDim.x + threadIdx.x;   // index in float4s
  if (i >= (size_t)n * (M / 4)) return;
  int c4 = (int)(i % (M / 4));
  float4 a = *(const float4*)(acc + i * 4);
  float4 b = *(const float4*)(bias + c4 * 4);
  float4 o = make_float4(lrelu(a.x + b.x, ACT_SLOPE), lrelu(a.y + b.y, ACT_SLOPE),
                         lrelu(a.z + b.z, ACT_SLOPE), lrelu(a.w + b.w, ACT_SLOPE));
  *(float4*)(out + i * 4) = o;
}
__global__ void k_fin_mean(const float* __restrict__ acc, const float* __restrict__ bias,
                           float* __restrict__ out, int n, int Cc, int act) {
  size_t i = (size_t)blockIdx.x * blockDim.x + threadIdx.x;
  if (i >= (size_t)n * Cc) return;
  int node = (int)(i / Cc), c = (int)(i % Cc);
  const float* p = acc + (size_t)node * 4 * Cc;
  float v = 0.25f * (p[c] + p[Cc + c] + p[2 * Cc + c] + p[3 * Cc + c]) + bias[c];
  out[i] = act ? lrelu(v, ACT_SLOPE) : v;
}

// =======================================================================
extern "C" void kernel_launch(void* const* d_in, const int* in_sizes, int n_in,
                              void* d_out, int out_size, void* d_ws, size_t ws_size,
                              hipStream_t stream) {
  const float* x  = (const float*)d_in[0];
  const int*   ei = (const int*)d_in[1];
  const int N = in_sizes[0] / 256;
  const int E = in_sizes[1] / 2;
  const int* src = ei;
  const int* dst = ei + E;

  const float* Wl[3]   = {(const float*)d_in[2],  (const float*)d_in[8],  (const float*)d_in[14]};
  const float* bl[3]   = {(const float*)d_in[3],  (const float*)d_in[9],  (const float*)d_in[15]};
  const float* Wr[3]   = {(const float*)d_in[4],  (const float*)d_in[10], (const float*)d_in[16]};
  const float* br[3]   = {(const float*)d_in[5],  (const float*)d_in[11], (const float*)d_in[17]};
  const float* att[3]  = {(const float*)d_in[6],  (const float*)d_in[12], (const float*)d_in[18]};
  const float* bias[3] = {(const float*)d_in[7],  (const float*)d_in[13], (const float*)d_in[19]};

  // ---- workspace carve (~250 MB) ----
  char* p = (char*)d_ws;
  auto carve = [&](size_t bytes) -> char* {
    char* r = p; p += (bytes + 255) & ~(size_t)255; return r;
  };
  float*  XL   = (float*)carve((size_t)N * 256 * 4);
  float*  XR   = (float*)carve((size_t)N * 256 * 4);
  float*  Hbuf = (float*)carve((size_t)N * 256 * 4);
  float*  ACC  = (float*)carve((size_t)N * 256 * 4);
  float*  LOG  = (float*)carve((size_t)E * 4 * 4);
  float*  MB   = (float*)carve((size_t)N * 4 * 4);
  float*  DB   = (float*)carve((size_t)N * 4 * 4);
  __bf16* ABF  = (__bf16*)carve((size_t)N * 256 * 2);
  __bf16* WPL  = (__bf16*)carve((size_t)8 * 16 * 512 * 2);
  __bf16* WPR  = (__bf16*)carve((size_t)8 * 16 * 512 * 2);
  int* ROWPTR  = (int*)carve((size_t)(N + 1) * 4);
  int* CURSOR  = (int*)carve((size_t)(N + 1) * 4);
  int* CSR     = (int*)carve((size_t)E * 4);

  float* z_out  = (float*)d_out;
  float* a1_out = z_out + (size_t)N * 10;
  float* a2_out = a1_out + (size_t)E * 4;
  float* az_out = a2_out + (size_t)E * 4;

  // ---- CSR of incoming edges (shared by all 3 layers; graph is static) ----
  hipMemsetAsync(CURSOR, 0, (size_t)N * 4, stream);
  k_count  <<<(E + 255) / 256, 256, 0, stream>>>(dst, CURSOR, E);
  k_scan   <<<1, 1024, 0, stream>>>(CURSOR, ROWPTR, N);
  k_copy_int<<<(N + 255) / 256, 256, 0, stream>>>(ROWPTR, CURSOR, N);
  k_scatter<<<(E + 255) / 256, 256, 0, stream>>>(dst, CURSOR, CSR, E);

  const int mtiles = N / 16;                       // 50000 -> 3125 (exact)
  const int gbx = (mtiles + 7) / 8;                // 8 waves / block
  const int ebx = (E + 7) / 8;
  const int nbx = (N + 7) / 8;

  // =========== layer 1: din=256, heads*cout=256, concat ===========
  {
    const int K = 256, M = 256, ntp = 16, pt = (K / 32) * ntp * 32;
    k_f32_to_bf16<<<(unsigned)(((size_t)N * 256 + 255) / 256), 256, 0, stream>>>(x, ABF, (size_t)N * 256);
    k_pack_w<<<(pt + 255) / 256, 256, 0, stream>>>(Wl[0], WPL, K, M, ntp);
    k_pack_w<<<(pt + 255) / 256, 256, 0, stream>>>(Wr[0], WPR, K, M, ntp);
    k_gemm_wmma<<<dim3(gbx, ntp / 4), 256, 0, stream>>>(ABF, WPL, bl[0], XL, N, K, M, ntp);
    k_gemm_wmma<<<dim3(gbx, ntp / 4), 256, 0, stream>>>(ABF, WPR, br[0], XR, N, K, M, ntp);
    k_edge_logits_v4<<<ebx, 256, 0, stream>>>(XL, XR, att[0], src, dst, LOG, E);
    k_softmax_stats<<<(N + 255) / 256, 256, 0, stream>>>(LOG, ROWPTR, CSR, MB, DB, N);
    k_aggregate_v4<<<nbx, 256, 0, stream>>>(XL, LOG, MB, DB, ROWPTR, CSR, src, a1_out, ACC, N);
    k_fin_concat4<<<(unsigned)(((size_t)N * 64 + 255) / 256), 256, 0, stream>>>(ACC, bias[0], Hbuf, N, 256);
  }
  // =========== layer 2: din=256, heads*cout=256, mean -> [N,64] ===========
  {
    const int K = 256, M = 256, ntp = 16, pt = (K / 32) * ntp * 32;
    k_f32_to_bf16<<<(unsigned)(((size_t)N * 256 + 255) / 256), 256, 0, stream>>>(Hbuf, ABF, (size_t)N * 256);
    k_pack_w<<<(pt + 255) / 256, 256, 0, stream>>>(Wl[1], WPL, K, M, ntp);
    k_pack_w<<<(pt + 255) / 256, 256, 0, stream>>>(Wr[1], WPR, K, M, ntp);
    k_gemm_wmma<<<dim3(gbx, ntp / 4), 256, 0, stream>>>(ABF, WPL, bl[1], XL, N, K, M, ntp);
    k_gemm_wmma<<<dim3(gbx, ntp / 4), 256, 0, stream>>>(ABF, WPR, br[1], XR, N, K, M, ntp);
    k_edge_logits_v4<<<ebx, 256, 0, stream>>>(XL, XR, att[1], src, dst, LOG, E);
    k_softmax_stats<<<(N + 255) / 256, 256, 0, stream>>>(LOG, ROWPTR, CSR, MB, DB, N);
    k_aggregate_v4<<<nbx, 256, 0, stream>>>(XL, LOG, MB, DB, ROWPTR, CSR, src, a2_out, ACC, N);
    k_fin_mean<<<(unsigned)(((size_t)N * 64 + 255) / 256), 256, 0, stream>>>(ACC, bias[1], Hbuf, N, 64, 1);
  }
  // =========== latent layer: din=64, heads*latent=40, mean -> z[N,10] ===========
  {
    const int K = 64, M = 40, ntp = 4, pt = (K / 32) * ntp * 32;
    k_f32_to_bf16<<<(unsigned)(((size_t)N * 64 + 255) / 256), 256, 0, stream>>>(Hbuf, ABF, (size_t)N * 64);
    k_pack_w<<<(pt + 255) / 256, 256, 0, stream>>>(Wl[2], WPL, K, M, ntp);
    k_pack_w<<<(pt + 255) / 256, 256, 0, stream>>>(Wr[2], WPR, K, M, ntp);
    k_gemm_wmma<<<dim3(gbx, 1), 256, 0, stream>>>(ABF, WPL, bl[2], XL, N, K, M, ntp);
    k_gemm_wmma<<<dim3(gbx, 1), 256, 0, stream>>>(ABF, WPR, br[2], XR, N, K, M, ntp);
    k_edge_logits<40, 10><<<ebx, 256, 0, stream>>>(XL, XR, att[2], src, dst, LOG, E);
    k_softmax_stats<<<(N + 255) / 256, 256, 0, stream>>>(LOG, ROWPTR, CSR, MB, DB, N);
    k_aggregate<40, 10><<<nbx, 256, 0, stream>>>(XL, LOG, MB, DB, ROWPTR, CSR, src, az_out, ACC, N);
    k_fin_mean<<<(unsigned)(((size_t)N * 10 + 255) / 256), 256, 0, stream>>>(ACC, bias[2], z_out, N, 10, 0);
  }
}
